// personal_attention_49228915147114
// MI455X (gfx1250) — matrix operand order repaired and checked by reading, not compiled, not run on hardware
//
#include <hip/hip_runtime.h>
#include <hip/hip_bf16.h>
#include <math.h>

// ---------------------------------------------------------------------------
// Types for CDNA5 WMMA (gfx1250, wave32)
// ---------------------------------------------------------------------------
typedef __attribute__((ext_vector_type(16))) __bf16        v16bf;
typedef __attribute__((ext_vector_type(8)))  float         v8f;
typedef __attribute__((ext_vector_type(4)))  unsigned int  u32x4;

union Frag { v16bf v; u32x4 u[2]; };

#define B_   64
#define C_   960
#define HW_  14
#define L_   196            // 14*14
#define BL_  12544          // 64*196
#define NH_  8
#define HD_  120            // 960/8
#define CH_  3840           // 4*C

__device__ __forceinline__ unsigned short f2bf(float f) {
  unsigned u = __float_as_uint(f);
  unsigned r = ((u >> 16) & 1u) + 0x7fffu;        // round-to-nearest-even
  return (unsigned short)((u + r) >> 16);
}

// gfx1250 async DMA: global -> LDS, 16B per lane, tracked by ASYNCcnt
__device__ __forceinline__ void async_ld16(unsigned lds_off, const void* gptr) {
  asm volatile("global_load_async_to_lds_b128 %0, %1, off"
               :: "v"(lds_off), "v"((unsigned long long)(size_t)gptr)
               : "memory");
}
__device__ __forceinline__ void wait_async0() {
  asm volatile("s_wait_asynccnt 0x0" ::: "memory");
}
__device__ __forceinline__ unsigned lds_off(const void* p) {
  return (unsigned)(size_t)p;   // low 32 bits of flat LDS-aperture addr = LDS offset
}

// ---------------------------------------------------------------------------
// WMMA bf16 GEMM:  out[M,N] = act( A[M,K] @ W[N,K]^T + bias ) (+resid)
//   macro tile 128(M) x 64(N), 256 threads = 8 waves, each wave 32x32.
//   Double-buffered LDS staged via global_load_async_to_lds_b128.
//   act: 0=none 1=silu 2=gelu(exact) 3=elu+1
//   trans: store outF as (b, n, l) with m = b*196+l   (final conv layout)
// ---------------------------------------------------------------------------
__global__ __launch_bounds__(256)
void k_gemm_bf16(const unsigned short* __restrict__ A,
                 const unsigned short* __restrict__ W,
                 const float* __restrict__ bias,
                 const float* __restrict__ resid,
                 float* __restrict__ outF,
                 unsigned short* __restrict__ outH,
                 int M, int N, int K, int act, int trans) {
  __shared__ unsigned short sA[2][128 * 40];   // 128 rows x 32 K, stride 40 (20 dwords)
  __shared__ unsigned short sB[2][64 * 40];

  const int tid   = threadIdx.x;
  const int lane  = tid & 31;
  const int wv    = tid >> 5;     // 0..7
  const int mW    = wv >> 1;      // 0..3 : 32-row strip
  const int nW    = wv & 1;       // 0..1 : 32-col strip
  const int ln    = lane & 15;
  const int sel   = lane >> 4;    // 0: lanes 0-15, 1: lanes 16-31

  const int mBase = blockIdx.y * 128;
  const int nBase = blockIdx.x * 64;

  const v8f vzero = {0.f,0.f,0.f,0.f,0.f,0.f,0.f,0.f};
  v8f acc[2][2] = {{vzero, vzero}, {vzero, vzero}};

  const int rA = tid >> 1, hA = tid & 1;          // A tile: 2 threads / row, 32B each
  const unsigned short* gA = A + (size_t)(mBase + rA) * K + hA * 16;
  const unsigned short* gB = (tid < 128)
      ? W + (size_t)(nBase + (tid >> 1)) * K + (tid & 1) * 16 : nullptr;

  // issue one K-slab (32 wide) of A+B tiles into LDS buffer `buf`
  auto issue = [&](int kb, int buf) {
    const unsigned short* ga = gA + kb;
    unsigned la = lds_off(&sA[buf][rA * 40 + hA * 16]);
    async_ld16(la,      ga);
    async_ld16(la + 16, ga + 8);
    if (tid < 128) {
      const unsigned short* gb = gB + kb;
      unsigned lb = lds_off(&sB[buf][(tid >> 1) * 40 + (tid & 1) * 16]);
      async_ld16(lb,      gb);
      async_ld16(lb + 16, gb + 8);
    }
    if (kb + 64 < K) __builtin_prefetch(ga + 64, 0, 1);   // global_prefetch_b8, 2 slabs ahead
  };

  const int nsteps = K >> 5;
  issue(0, 0);
  for (int s = 0; s < nsteps; ++s) {
    const int buf = s & 1;
    wait_async0();        // my async writes into `buf` have landed
    __syncthreads();      // everyone's writes into `buf` visible
    if (s + 1 < nsteps) issue((s + 1) << 5, buf ^ 1);

    // B fragments: lane<16 -> K0..15 (col N=ln), lane>=16 -> K16..31
    Frag bf0, bf1;
    {
      const u32x4* p0 = (const u32x4*)&sB[buf][(nW * 32 + 0 * 16 + ln) * 40 + sel * 16];
      bf0.u[0] = p0[0]; bf0.u[1] = p0[1];
      const u32x4* p1 = (const u32x4*)&sB[buf][(nW * 32 + 1 * 16 + ln) * 40 + sel * 16];
      bf1.u[0] = p1[0]; bf1.u[1] = p1[1];
    }
#pragma unroll
    for (int ms = 0; ms < 2; ++ms) {
      // A fragment: lane<16 -> K0..15; lane>=16 -> K16..31
      Frag af;
      const unsigned short* pa = &sA[buf][(mW * 32 + ms * 16 + ln) * 40];
      af.u[0] = *(const u32x4*)(pa + sel * 16);
      af.u[1] = *(const u32x4*)(pa + sel * 16 + 8);
      acc[ms][0] = __builtin_amdgcn_wmma_f32_16x16x32_bf16(
          false, af.v, false, bf0.v, (short)0, acc[ms][0], false, false);
      acc[ms][1] = __builtin_amdgcn_wmma_f32_16x16x32_bf16(
          false, af.v, false, bf1.v, (short)0, acc[ms][1], false, false);
    }
    __syncthreads();      // all reads of `buf` done before step s+2 overwrites it
  }

  // epilogue: C/D layout -> lane<16: (M=r, N=ln); lane>=16: (M=8+r, N=ln)
#pragma unroll
  for (int ms = 0; ms < 2; ++ms) {
#pragma unroll
    for (int ns = 0; ns < 2; ++ns) {
      const int nCol = nBase + nW * 32 + ns * 16 + ln;
      const int mRow = mBase + mW * 32 + ms * 16 + sel * 8;
      const float bv = bias ? bias[nCol] : 0.f;
      v8f a = acc[ms][ns];
#pragma unroll
      for (int r = 0; r < 8; ++r) {
        const int m = mRow + r;
        float v = a[r] + bv;
        if      (act == 1) v = v / (1.f + __expf(-v));                       // silu
        else if (act == 2) v = 0.5f * v * (1.f + erff(v * 0.70710678118f));  // gelu exact
        else if (act == 3) v = (v > 0.f) ? v + 1.f : __expf(v);              // elu+1
        if (resid) v += resid[(size_t)m * N + nCol];
        if (outF) {
          if (trans) {
            int b = m / L_, l = m - b * L_;
            outF[((size_t)b * N + nCol) * L_ + l] = v;
          } else {
            outF[(size_t)m * N + nCol] = v;
          }
        }
        if (outH) outH[(size_t)m * N + nCol] = f2bf(v);
      }
    }
  }
}

// ---------------------------------------------------------------------------
// (B,C,H,W) f32 -> (B,L,C) f32
// ---------------------------------------------------------------------------
__global__ void k_nchw_to_blc(const float* __restrict__ in, float* __restrict__ out) {
  int idx = blockIdx.x * 256 + threadIdx.x;          // over B*L*C
  int c = idx % C_;
  int l = (idx / C_) % L_;
  int b = idx / (C_ * L_);
  out[idx] = in[((size_t)b * C_ + c) * L_ + l];
}

// ---------------------------------------------------------------------------
// f32 -> bf16 elementwise
// ---------------------------------------------------------------------------
__global__ void k_f2bf(const float* __restrict__ src, unsigned short* __restrict__ dst, int n) {
  int i = blockIdx.x * 256 + threadIdx.x;
  if (i < n) dst[i] = f2bf(src[i]);
}

// ---------------------------------------------------------------------------
// depthwise 3x3 (SAME) on (B,L,C):
//  mode 0 (cpe):  outF = in + conv(in) + b
//  mode 1 (dwc):  t = silu(conv(in)+b); outF = t; outH = bf16(t)
//  mode 2 (lepe): t = (addSrc + conv(in) + b) * mulSrc; outH = bf16(t)
// ---------------------------------------------------------------------------
__global__ void k_dwconv(const float* __restrict__ in, const float* __restrict__ w,
                         const float* __restrict__ bias,
                         const float* __restrict__ addSrc, const float* __restrict__ mulSrc,
                         float* __restrict__ outF, unsigned short* __restrict__ outH,
                         int mode) {
  int idx = blockIdx.x * 256 + threadIdx.x;          // over B*L*C
  int c = idx % C_;
  int l = (idx / C_) % L_;
  int b = idx / (C_ * L_);
  int hh = l / HW_, ww = l - hh * HW_;
  float s = 0.f;
#pragma unroll
  for (int kh = 0; kh < 3; ++kh) {
#pragma unroll
    for (int kw = 0; kw < 3; ++kw) {
      int h2 = hh + kh - 1, w2 = ww + kw - 1;
      if (h2 >= 0 && h2 < HW_ && w2 >= 0 && w2 < HW_) {
        s += in[((size_t)b * L_ + h2 * HW_ + w2) * C_ + c] * w[c * 9 + kh * 3 + kw];
      }
    }
  }
  s += bias[c];
  if (mode == 0) {
    outF[idx] = in[idx] + s;
  } else if (mode == 1) {
    float t = s / (1.f + __expf(-s));
    outF[idx] = t;
    outH[idx] = f2bf(t);
  } else {
    float t = (addSrc[idx] + s) * mulSrc[idx];
    outH[idx] = f2bf(t);
  }
}

// ---------------------------------------------------------------------------
// LayerNorm over C=960, one block per row, writes bf16
// ---------------------------------------------------------------------------
__global__ __launch_bounds__(256)
void k_ln(const float* __restrict__ x, const float* __restrict__ g,
          const float* __restrict__ b, unsigned short* __restrict__ outH) {
  __shared__ float rs[8], rs2[8], stat[2];
  const int row = blockIdx.x;
  const int tid = threadIdx.x;
  float s = 0.f, s2 = 0.f;
  for (int c = tid; c < C_; c += 256) {
    float v = x[(size_t)row * C_ + c];
    s += v; s2 += v * v;
  }
  for (int o = 16; o > 0; o >>= 1) { s += __shfl_down(s, o); s2 += __shfl_down(s2, o); }
  if ((tid & 31) == 0) { rs[tid >> 5] = s; rs2[tid >> 5] = s2; }
  __syncthreads();
  if (tid == 0) {
    float a = 0.f, q = 0.f;
    for (int i = 0; i < 8; ++i) { a += rs[i]; q += rs2[i]; }
    float m = a / C_;
    stat[0] = m;
    stat[1] = rsqrtf(q / C_ - m * m + 1e-5f);
  }
  __syncthreads();
  float m = stat[0], inv = stat[1];
  for (int c = tid; c < C_; c += 256) {
    float v = (x[(size_t)row * C_ + c] - m) * inv * g[c] + b[c];
    outH[(size_t)row * C_ + c] = f2bf(v);
  }
}

// ---------------------------------------------------------------------------
// kmean[b,c] = mean_l k[b,l,c]   (k = qk[..., 960+c], pre-RoPE)
// ---------------------------------------------------------------------------
__global__ void k_kmean(const float* __restrict__ qk, float* __restrict__ km) {
  int idx = blockIdx.x * 256 + threadIdx.x;          // over B*C
  if (idx >= B_ * C_) return;
  int b = idx / C_, c = idx - b * C_;
  float s = 0.f;
  for (int l = 0; l < L_; ++l)
    s += qk[((size_t)b * L_ + l) * (2 * C_) + C_ + c];
  km[idx] = s * (1.f / (float)L_);
}

// ---------------------------------------------------------------------------
// z[b,h,l] = 1 / (q[b,l,h,:].km[b,h,:] + 1e-6)   (pre-RoPE q)
// ---------------------------------------------------------------------------
__global__ void k_z(const float* __restrict__ qk, const float* __restrict__ km,
                    float* __restrict__ z) {
  int idx = blockIdx.x * 256 + threadIdx.x;          // over B*L*NH
  if (idx >= B_ * L_ * NH_) return;
  int h = idx % NH_;
  int l = (idx / NH_) % L_;
  int b = idx / (NH_ * L_);
  float s = 0.f;
  const float* qp  = qk + ((size_t)b * L_ + l) * (2 * C_) + h * HD_;
  const float* kmp = km + (size_t)b * C_ + h * HD_;
  for (int d = 0; d < HD_; ++d) s += qp[d] * kmp[d];
  z[((size_t)b * NH_ + h) * L_ + l] = 1.f / (s + 1e-6f);
}

// ---------------------------------------------------------------------------
// 2D RoPE, in-place on qk (both q and k halves); pair p of 480 per token
// ---------------------------------------------------------------------------
__global__ void k_rope(float* __restrict__ qk) {
  int idx = blockIdx.x * 256 + threadIdx.x;          // over 2*B*L*480
  int p = idx % (C_ / 2);
  int l = (idx / (C_ / 2)) % L_;
  int rest = idx / ((C_ / 2) * L_);
  int b = rest % B_;
  int half = rest / B_;                              // 0=q, 1=k
  int hh = l / HW_, ww = l - hh * HW_;
  const int kmax = C_ / 4;                            // 240
  int j   = (p < kmax) ? p : p - kmax;
  int pos = (p < kmax) ? hh : ww;
  float theta = __expf(-(float)j * (9.210340371976184f / (float)kmax)); // 10000^{-j/kmax}
  float ang = (float)pos * theta;
  float sn, cs;
  __sincosf(ang, &sn, &cs);
  size_t base = ((size_t)b * L_ + l) * (2 * C_) + half * C_ + 2 * p;
  float re = qk[base], im = qk[base + 1];
  qk[base]     = cs * re - sn * im;
  qk[base + 1] = sn * re + cs * im;
}

// ---------------------------------------------------------------------------
// kv[b,h,d,e] = (1/L) * sum_l k_r[b,l,h,d] * v[b,l,h,e]   (one WG per (b,h))
// ---------------------------------------------------------------------------
__global__ __launch_bounds__(256)
void k_kv(const float* __restrict__ qk, const float* __restrict__ v,
          float* __restrict__ kv) {
  const int b = blockIdx.x >> 3, h = blockIdx.x & 7;
  __shared__ float sk[28 * HD_];
  __shared__ float sv[28 * HD_];
  float acc[57];
#pragma unroll
  for (int j = 0; j < 57; ++j) acc[j] = 0.f;
  for (int lc = 0; lc < L_; lc += 28) {
    __syncthreads();
    for (int i = threadIdx.x; i < 28 * HD_; i += 256) {
      int li = i / HD_, d = i - li * HD_;
      size_t row = (size_t)(b * L_ + lc + li);
      sk[i] = qk[row * (2 * C_) + C_ + h * HD_ + d];
      sv[i] = v[row * C_ + h * HD_ + d];
    }
    __syncthreads();
    for (int j = 0; j < 57; ++j) {
      int idx = threadIdx.x + j * 256;
      if (idx < HD_ * HD_) {
        int d = idx / HD_, e = idx - d * HD_;
        float a = acc[j];
        for (int li = 0; li < 28; ++li) a += sk[li * HD_ + d] * sv[li * HD_ + e];
        acc[j] = a;
      }
    }
  }
  const float sc = 1.f / (float)L_;
  for (int j = 0; j < 57; ++j) {
    int idx = threadIdx.x + j * 256;
    if (idx < HD_ * HD_) kv[(size_t)blockIdx.x * (HD_ * HD_) + idx] = acc[j] * sc;
  }
}

// ---------------------------------------------------------------------------
// out[b,l,h,e] = z[b,h,l] * sum_d q_r[b,l,h,d] * kv[b,h,d,e]  (WG per (b,h))
// ---------------------------------------------------------------------------
__global__ __launch_bounds__(256)
void k_attnout(const float* __restrict__ qk, const float* __restrict__ kv,
               const float* __restrict__ z, float* __restrict__ out) {
  const int b = blockIdx.x >> 3, h = blockIdx.x & 7;
  __shared__ float skv[40 * HD_];    // kv rows d-chunk
  __shared__ float sq[L_ * 40];      // q[l][d-chunk]
  float acc[92];
#pragma unroll
  for (int j = 0; j < 92; ++j) acc[j] = 0.f;
  for (int dc = 0; dc < HD_; dc += 40) {
    __syncthreads();
    for (int i = threadIdx.x; i < 40 * HD_; i += 256) {
      int d = dc + i / HD_;
      skv[i] = kv[(size_t)blockIdx.x * (HD_ * HD_) + d * HD_ + (i % HD_)];
    }
    for (int i = threadIdx.x; i < L_ * 40; i += 256) {
      int l = i / 40, d = dc + (i - l * 40);
      sq[i] = qk[((size_t)b * L_ + l) * (2 * C_) + h * HD_ + d];
    }
    __syncthreads();
    for (int j = 0; j < 92; ++j) {
      int idx = threadIdx.x + j * 256;
      if (idx < L_ * HD_) {
        int l = idx / HD_, e = idx - l * HD_;
        float a = acc[j];
        for (int d = 0; d < 40; ++d) a += sq[l * 40 + d] * skv[d * HD_ + e];
        acc[j] = a;
      }
    }
  }
  for (int j = 0; j < 92; ++j) {
    int idx = threadIdx.x + j * 256;
    if (idx < L_ * HD_) {
      int l = idx / HD_, e = idx - l * HD_;
      out[((size_t)b * L_ + l) * C_ + h * HD_ + e] =
          acc[j] * z[((size_t)b * NH_ + h) * L_ + l];
    }
  }
}

// ---------------------------------------------------------------------------
// host side
// ---------------------------------------------------------------------------
extern "C" void kernel_launch(void* const* d_in, const int* in_sizes, int n_in,
                              void* d_out, int out_size, void* d_ws, size_t ws_size,
                              hipStream_t stream) {
  (void)in_sizes; (void)n_in; (void)out_size; (void)ws_size;

  const float* x      = (const float*)d_in[0];
  const float* cpe1_w = (const float*)d_in[1];
  const float* cpe1_b = (const float*)d_in[2];
  const float* n1_g   = (const float*)d_in[3];
  const float* n1_b   = (const float*)d_in[4];
  const float* in_w   = (const float*)d_in[5];
  const float* in_b   = (const float*)d_in[6];
  const float* act_w  = (const float*)d_in[7];
  const float* act_b  = (const float*)d_in[8];
  const float* dwc_w  = (const float*)d_in[9];
  const float* dwc_b  = (const float*)d_in[10];
  const float* qk_w   = (const float*)d_in[11];
  const float* qk_b   = (const float*)d_in[12];
  const float* lepe_w = (const float*)d_in[13];
  const float* lepe_b = (const float*)d_in[14];
  const float* out_w  = (const float*)d_in[15];
  const float* out_b  = (const float*)d_in[16];
  const float* cpe2_w = (const float*)d_in[17];
  const float* cpe2_b = (const float*)d_in[18];
  const float* n2_g   = (const float*)d_in[19];
  const float* n2_b   = (const float*)d_in[20];
  const float* fc1_w  = (const float*)d_in[21];
  const float* fc1_b  = (const float*)d_in[22];
  const float* fc2_w  = (const float*)d_in[23];
  const float* fc2_b  = (const float*)d_in[24];
  const float* conv_w = (const float*)d_in[25];
  const float* conv_b = (const float*)d_in[26];

  const size_t BLC = (size_t)BL_ * C_;            // 12,042,240
  const size_t S   = BLC * 4;                     // f32 buffer bytes

  unsigned char* ws = (unsigned char*)d_ws;
  float* F0   = (float*)(ws + 0 * S);             // x
  float* F1   = (float*)(ws + 1 * S);             // cpe out / shortcut
  float* F2   = (float*)(ws + 2 * S);             // gemm f32 temp / attn-out
  float* F3   = (float*)(ws + 3 * S);             // act_res
  float* F4   = (float*)(ws + 4 * S);             // xi (v)
  float* RBIG = (float*)(ws + 5 * S);             // 2*S: qk f32 | fc1-out bf16
  unsigned short* HA = (unsigned short*)(ws + 7 * S);                 // bf16 BLC
  float* FKV  = (float*)(ws + 7 * S + BLC * 2);                       // 512*14400 f32
  float* FZ   = (float*)((unsigned char*)FKV + (size_t)512 * HD_ * HD_ * 4);
  float* FKM  = (float*)((unsigned char*)FZ + (size_t)B_ * NH_ * L_ * 4);
  unsigned short* HW = (unsigned short*)((unsigned char*)FKM + (size_t)B_ * C_ * 4);

  // bf16 weight offsets (elements)
  unsigned short* W_act  = HW;
  unsigned short* W_in   = W_act + (size_t)C_ * C_;
  unsigned short* W_qk   = W_in  + (size_t)C_ * C_;
  unsigned short* W_out  = W_qk  + (size_t)2 * C_ * C_;
  unsigned short* W_fc1  = W_out + (size_t)C_ * C_;
  unsigned short* W_fc2  = W_fc1 + (size_t)CH_ * C_;
  unsigned short* W_conv = W_fc2 + (size_t)C_ * CH_;

  const dim3 blk(256);
  auto cvt = [&](const float* src, unsigned short* dst, int n) {
    k_f2bf<<<dim3((n + 255) / 256), blk, 0, stream>>>(src, dst, n);
  };
  // one-time weight conversion (deterministic; redone every launch)
  cvt(act_w,  W_act,  C_ * C_);
  cvt(in_w,   W_in,   C_ * C_);
  cvt(qk_w,   W_qk,   2 * C_ * C_);
  cvt(out_w,  W_out,  C_ * C_);
  cvt(fc1_w,  W_fc1,  CH_ * C_);
  cvt(fc2_w,  W_fc2,  C_ * CH_);
  cvt(conv_w, W_conv, 640 * C_);

  const int gEl = (int)(BLC / 256);               // 47040, exact
  k_nchw_to_blc<<<dim3(gEl), blk, 0, stream>>>(x, F0);

  auto gemm = [&](const unsigned short* A, const unsigned short* Wm,
                  const float* bias, const float* resid,
                  float* oF, unsigned short* oH, int N, int K, int act, int trans) {
    k_gemm_bf16<<<dim3(N / 64, BL_ / 128), blk, 0, stream>>>(
        A, Wm, bias, resid, oF, oH, BL_, N, K, act, trans);
  };

  for (int it = 0; it < 3; ++it) {
    // cpe1: F1 = F0 + dw(F0)+b
    k_dwconv<<<dim3(gEl), blk, 0, stream>>>(F0, cpe1_w, cpe1_b, nullptr, nullptr, F1, nullptr, 0);
    // LN1 -> HA (bf16)
    k_ln<<<dim3(BL_), blk, 0, stream>>>(F1, n1_g, n1_b, HA);
    // act_res = silu(xn @ act_w^T + b) -> F3
    gemm(HA, W_act, act_b, nullptr, F3, nullptr, C_, C_, 1, 0);
    // xi = xn @ in_w^T + b -> F2
    gemm(HA, W_in, in_b, nullptr, F2, nullptr, C_, C_, 0, 0);
    // xi = silu(dw(xi)+b) -> F4 (f32) + HA (bf16)
    k_dwconv<<<dim3(gEl), blk, 0, stream>>>(F2, dwc_w, dwc_b, nullptr, nullptr, F4, HA, 1);
    // qk = elu(xi @ qk_w^T + b)+1 -> RBIG (B,L,2C) f32
    gemm(HA, W_qk, qk_b, nullptr, RBIG, nullptr, 2 * C_, C_, 3, 0);
    // normalizer from pre-RoPE q/k
    k_kmean<<<dim3((B_ * C_ + 255) / 256), blk, 0, stream>>>(RBIG, FKM);
    k_z<<<dim3((B_ * L_ * NH_ + 255) / 256), blk, 0, stream>>>(RBIG, FKM, FZ);
    // RoPE in place
    k_rope<<<dim3(gEl), blk, 0, stream>>>(RBIG);
    // kv and attention output
    k_kv<<<dim3(B_ * NH_), blk, 0, stream>>>(RBIG, F4, FKV);
    k_attnout<<<dim3(B_ * NH_), blk, 0, stream>>>(RBIG, FKV, FZ, F2);
    // xa = (attn + lepe(v)) * act_res -> HA (bf16)
    k_dwconv<<<dim3(gEl), blk, 0, stream>>>(F4, lepe_w, lepe_b, F2, F3, nullptr, HA, 2);
    // x = shortcut + xa @ out_w^T + b -> F2
    gemm(HA, W_out, out_b, F1, F2, nullptr, C_, C_, 0, 0);
    // cpe2: F1 = F2 + dw(F2)+b
    k_dwconv<<<dim3(gEl), blk, 0, stream>>>(F2, cpe2_w, cpe2_b, nullptr, nullptr, F1, nullptr, 0);
    // LN2 -> HA
    k_ln<<<dim3(BL_), blk, 0, stream>>>(F1, n2_g, n2_b, HA);
    // h = gelu(xn @ fc1^T + b) -> RBIG as bf16
    gemm(HA, W_fc1, fc1_b, nullptr, nullptr, (unsigned short*)RBIG, CH_, C_, 2, 0);
    // x = F1 + h @ fc2^T + b -> F0 (next iteration input)
    gemm((unsigned short*)RBIG, W_fc2, fc2_b, F1, F0, nullptr, C_, CH_, 0, 0);
  }

  // final 1x1 conv: (B,L,C) @ conv_w^T -> d_out as (B,640,H,W)
  cvt(F0, HA, (int)BLC);
  gemm(HA, W_conv, conv_b, nullptr, (float*)d_out, nullptr, 640, C_, 0, 1);
}
